// BinarizedConv2d_57501022158954
// MI455X (gfx1250) — matrix operand order repaired and checked by reading, not compiled, not run on hardware
//
#include <hip/hip_runtime.h>

// activation[d,o] = sum_k W[d,o,k] * x[d,k]  (all values are 0/1 floats)
// out[d,o] = (activation > bias[d,o]) ? 1.0 : 0.0
//
// D=128, O=256, K=2304. 302MB of f32 weights, read-once -> pure HBM-BW bound
// (~13us floor at 23.3 TB/s). Strategy: stream W with non-temporal b128 loads,
// convert f32->f16 (exact for 0/1), feed v_wmma_f32_16x16x32_f16 with x
// broadcast across the 16 B-columns. K split 4-ways across the block's waves
// for in-flight-load depth (8192 waves total), LDS reduction, then compare.

typedef __attribute__((ext_vector_type(16))) _Float16      v16h;
typedef __attribute__((ext_vector_type(8)))  float         v8f;
typedef __attribute__((ext_vector_type(4)))  float         v4f;
typedef __attribute__((ext_vector_type(16))) unsigned char v16ub;

#define D_DIRS 128
#define O_CH   256
#define K_LEN  2304
#define WAVES  4
#define K_PER_WAVE (K_LEN / WAVES)   // 576
#define K_ITERS    (K_PER_WAVE / 32) // 18

__global__ __launch_bounds__(WAVES * 32)
void binconv_wmma_kernel(const float* __restrict__ w,          // [D, O, K]
                         const float* __restrict__ bias,       // [D, O]
                         const unsigned char* __restrict__ x,  // [D, K] bool bytes
                         float* __restrict__ out)              // [D, O]
{
    const int lane = threadIdx.x & 31;
    const int wave = threadIdx.x >> 5;
    const int og   = blockIdx.x;   // output-channel tile: 0..15
    const int d    = blockIdx.y;   // direction: 0..127

    const int row  = lane & 15;    // M within the 16x16 tile
    const int half = lane >> 4;    // lane-group: selects K sub-block per ISA layout

    // Row base for this lane's A-matrix row (M = og*16 + row)
    const float* wrow = w + ((size_t)d * O_CH + (size_t)og * 16 + row) * K_LEN;
    const unsigned char* xrow = x + (size_t)d * K_LEN;

    int k0 = wave * K_PER_WAVE;
    v8f acc = {};

#pragma unroll 2
    for (int it = 0; it < K_ITERS; ++it, k0 += 32) {
        // ---- A: 16x32 f32 tile of W, converted to f16 (exact for 0/1).
        // A 16x32 layout (contiguous K order, matching B): lanes 0-15 hold K {0-15} of row M,
        // lanes 16-31 hold K {16-31}. Non-temporal: W is read-once and
        // larger than L2; don't evict the cached x/bias broadcast lines.
        const float* ap = wrow + k0 + half * 16;
        v4f a0 = __builtin_nontemporal_load((const v4f*)(ap + 0));
        v4f a1 = __builtin_nontemporal_load((const v4f*)(ap + 4));
        v4f a2 = __builtin_nontemporal_load((const v4f*)(ap + 8));
        v4f a3 = __builtin_nontemporal_load((const v4f*)(ap + 12));
        v16h a;
#pragma unroll
        for (int i = 0; i < 4; ++i) {
            a[i]      = (_Float16)a0[i];
            a[4 + i]  = (_Float16)a1[i];
            a[8 + i]  = (_Float16)a2[i];
            a[12 + i] = (_Float16)a3[i];
        }

        // ---- B: x[k0 + half*16 .. +16) replicated into all 16 columns.
        // B layout: VGPR i, lane L holds K = 2i + (L<16?0:16) (+1), N = L%16;
        // every lane in a half-group loads the same 16 bytes -> L0/L2 hit.
        v16ub xb = *(const v16ub*)(xrow + k0 + half * 16);
        v16h b;
#pragma unroll
        for (int i = 0; i < 16; ++i)
            b[i] = xb[i] ? (_Float16)1.0f : (_Float16)0.0f;

        acc = __builtin_amdgcn_wmma_f32_16x16x32_f16(
            /*neg_a=*/false, a, /*neg_b=*/false, b,
            /*c_mod=*/(short)0, acc, /*reuse_a=*/false, /*reuse_b=*/false);
    }

    // ---- Cross-wave K reduction through LDS ----
    __shared__ float red[WAVES - 1][32][8];
    if (wave != 0) {
#pragma unroll
        for (int r = 0; r < 8; ++r) red[wave - 1][lane][r] = acc[r];
    }
    __syncthreads();

    if (wave == 0) {
#pragma unroll
        for (int v = 0; v < WAVES - 1; ++v)
#pragma unroll
            for (int r = 0; r < 8; ++r) acc[r] += red[v][lane][r];

        // C/D layout: VGPR r, lanes 0-15 -> M=r, N=lane; lanes 16-31 -> M=8+r.
        // All N columns identical (B replicated); lanes with N==0 write out.
        if ((lane & 15) == 0) {
            const int obase = og * 16 + half * 8;
            const float* bp = bias + (size_t)d * O_CH + obase;
            float*       op = out  + (size_t)d * O_CH + obase;
#pragma unroll
            for (int r = 0; r < 8; ++r)
                op[r] = (acc[r] > bp[r]) ? 1.0f : 0.0f;
        }
    }
}

extern "C" void kernel_launch(void* const* d_in, const int* in_sizes, int n_in,
                              void* d_out, int out_size, void* d_ws, size_t ws_size,
                              hipStream_t stream) {
    (void)in_sizes; (void)n_in; (void)out_size; (void)d_ws; (void)ws_size;
    const float*         w    = (const float*)d_in[0];         // weight_noise [D,O,K] f32
    const float*         bias = (const float*)d_in[1];         // bias_noise   [D,O]   f32
    const unsigned char* x    = (const unsigned char*)d_in[2]; // x            [D,K]   bool bytes
    float*               out  = (float*)d_out;                 // [D,O] as 0.0/1.0

    dim3 grid(O_CH / 16, D_DIRS);   // 16 o-tiles x 128 directions = 2048 blocks
    dim3 block(WAVES * 32);         // 4 waves: K split 4-ways per block
    binconv_wmma_kernel<<<grid, block, 0, stream>>>(w, bias, x, out);
}
